// Cluster_12223476924713
// MI455X (gfx1250) — compile-verified, matrix operand order checked by reading, not run on hardware
//
#include <hip/hip_runtime.h>

#define N_ROWS 32768
#define K_CENT 1024
#define D_DIM  512
#define M_BLK  64      // rows per workgroup
#define KC     32      // bf16 k elements per WMMA chunk
#define KPAD   40      // padded LDS row (elements) -> 80B, 16B aligned, bank-spread
#define WAVES  16
#define THREADS (WAVES * 32)

typedef __attribute__((ext_vector_type(16))) __bf16 v16bf;
typedef __attribute__((ext_vector_type(8)))  float  v8f;

union Frag {
    uint4 q[2];
    v16bf v;
};
union Pack4 {
    __bf16 b[4];
    uint2  u;
};

// ---------------------------------------------------------------------------
// Prep: centers -> bf16 hi/lo planes + ||c||^2. One wave per center row.
// ---------------------------------------------------------------------------
__global__ void prep_centers(const float* __restrict__ centers,
                             __bf16* __restrict__ ch, __bf16* __restrict__ cl,
                             float* __restrict__ c2) {
    int wave = (int)((blockIdx.x * blockDim.x + threadIdx.x) >> 5);
    int lane = threadIdx.x & 31;
    if (wave >= K_CENT) return;
    const float* row = centers + (size_t)wave * D_DIM;
    float ss = 0.f;
    for (int i = 0; i < D_DIM; i += 32) {
        float x = row[lane + i];
        ss += x * x;
        __bf16 h = (__bf16)x;
        __bf16 l = (__bf16)(x - (float)h);
        ch[(size_t)wave * D_DIM + lane + i] = h;
        cl[(size_t)wave * D_DIM + lane + i] = l;
    }
    for (int m = 16; m >= 1; m >>= 1) ss += __shfl_xor(ss, m, 32);
    if (lane == 0) c2[wave] = ss;
}

// ---------------------------------------------------------------------------
// Prep: ||x||^2 per feature row. One wave per row.
// ---------------------------------------------------------------------------
__global__ void prep_x2(const float* __restrict__ f, float* __restrict__ x2) {
    int wave = (int)((blockIdx.x * blockDim.x + threadIdx.x) >> 5);
    int lane = threadIdx.x & 31;
    const float* row = f + (size_t)wave * D_DIM;
    float ss = 0.f;
    for (int i = 0; i < D_DIM; i += 32) {
        float x = row[lane + i];
        ss += x * x;
    }
    for (int m = 16; m >= 1; m >>= 1) ss += __shfl_xor(ss, m, 32);
    if (lane == 0) x2[wave] = ss;
}

// ---------------------------------------------------------------------------
// Main fused kernel: split-bf16 WMMA GEMM + distance + 1/d + row-normalize.
// Block = 64 rows x 1024 cols; 16 waves; wave w owns cols [w*64, w*64+64).
// ---------------------------------------------------------------------------
__global__ void __launch_bounds__(THREADS, 1)
fuzzy_mu_kernel(const float*  __restrict__ features,
                const __bf16* __restrict__ ch, const __bf16* __restrict__ cl,
                const float*  __restrict__ x2g, const float* __restrict__ c2g,
                float* __restrict__ out) {
    __shared__ __align__(16) __bf16 sAh[M_BLK * KPAD];
    __shared__ __align__(16) __bf16 sAl[M_BLK * KPAD];
    __shared__ float s_c2[K_CENT];
    __shared__ float s_x2[M_BLK];
    __shared__ float s_part[WAVES][M_BLK];
    __shared__ float s_scale[M_BLK];

    const int tid  = threadIdx.x;
    const int w    = tid >> 5;
    const int lane = tid & 31;
    const int grp  = lane >> 4;   // 0: lanes 0-15, 1: lanes 16-31
    const int r    = lane & 15;
    const int row0 = blockIdx.x * M_BLK;

    for (int i = tid; i < K_CENT; i += THREADS) s_c2[i] = c2g[i];
    if (tid < M_BLK) s_x2[tid] = x2g[row0 + tid];

    v8f acc[4][4] = {};

    // A staging assignment: thread handles 4 consecutive floats of a row
    const int arow = tid >> 3;          // 0..63
    const int acol = (tid & 7) * 4;     // 0,4,...,28

    for (int kc = 0; kc < D_DIM; kc += KC) {
        __syncthreads();  // previous iteration's fragment reads done

        // --- stage features f32 -> (hi, lo) bf16 in LDS ---
        float4 f4 = *(const float4*)(features + (size_t)(row0 + arow) * D_DIM + kc + acol);
        Pack4 ph, pl;
        {
            float xs[4] = {f4.x, f4.y, f4.z, f4.w};
            #pragma unroll
            for (int e = 0; e < 4; ++e) {
                __bf16 h = (__bf16)xs[e];
                ph.b[e] = h;
                pl.b[e] = (__bf16)(xs[e] - (float)h);
            }
        }
        *(uint2*)(sAh + arow * KPAD + acol) = ph.u;
        *(uint2*)(sAl + arow * KPAD + acol) = pl.u;
        __syncthreads();

        // --- A fragments (16x32 bf16 layout: grp selects K half-pattern) ---
        Frag ah[4], al[4];
        #pragma unroll
        for (int mt = 0; mt < 4; ++mt) {
            const char* p = (const char*)(sAh + (mt * 16 + r) * KPAD) + grp * 16;
            ah[mt].q[0] = *(const uint4*)p;
            ah[mt].q[1] = *(const uint4*)(p + 32);
            const char* q = (const char*)(sAl + (mt * 16 + r) * KPAD) + grp * 16;
            al[mt].q[0] = *(const uint4*)q;
            al[mt].q[1] = *(const uint4*)(q + 32);
        }

        // --- B fragments straight from L2 (centers bf16, row-major [K][D]) ---
        #pragma unroll
        for (int nt = 0; nt < 4; ++nt) {
            const int col = w * 64 + nt * 16 + r;
            const char* pbh = (const char*)(ch + (size_t)col * D_DIM + kc) + grp * 16;
            const char* pbl = (const char*)(cl + (size_t)col * D_DIM + kc) + grp * 16;
            Frag bh, bl;
            bh.q[0] = *(const uint4*)pbh;
            bh.q[1] = *(const uint4*)(pbh + 32);
            bl.q[0] = *(const uint4*)pbl;
            bl.q[1] = *(const uint4*)(pbl + 32);

            #pragma unroll
            for (int mt = 0; mt < 4; ++mt) {
                // split-bf16 fp32 emulation: hi*hi + hi*lo + lo*hi
                acc[mt][nt] = __builtin_amdgcn_wmma_f32_16x16x32_bf16(
                    false, ah[mt].v, false, bh.v, (short)0, acc[mt][nt], false, false);
                acc[mt][nt] = __builtin_amdgcn_wmma_f32_16x16x32_bf16(
                    false, ah[mt].v, false, bl.v, (short)0, acc[mt][nt], false, false);
                acc[mt][nt] = __builtin_amdgcn_wmma_f32_16x16x32_bf16(
                    false, al[mt].v, false, bh.v, (short)0, acc[mt][nt], false, false);
            }
        }
    }

    // --- epilogue: d = x2 - 2*dot + c2 ; inv = 1/d ; per-lane row partials ---
    float rowp[4][8];
    #pragma unroll
    for (int mt = 0; mt < 4; ++mt)
        #pragma unroll
        for (int j = 0; j < 8; ++j) rowp[mt][j] = 0.f;

    #pragma unroll
    for (int mt = 0; mt < 4; ++mt) {
        const int mbase = mt * 16 + grp * 8;
        #pragma unroll
        for (int nt = 0; nt < 4; ++nt) {
            const int col = w * 64 + nt * 16 + r;
            const float cc = s_c2[col];
            #pragma unroll
            for (int j = 0; j < 8; ++j) {
                float d  = s_x2[mbase + j] - 2.f * acc[mt][nt][j] + cc;
                float rc = __builtin_amdgcn_rcpf(d);
                rc = rc * (2.f - d * rc);      // one Newton step -> ~fp32 exact
                acc[mt][nt][j] = rc;
                rowp[mt][j] += rc;
            }
        }
    }

    // reduce across the 16 lanes of each half-wave (xor masks stay in-group)
    #pragma unroll
    for (int mt = 0; mt < 4; ++mt) {
        #pragma unroll
        for (int j = 0; j < 8; ++j) {
            float v = rowp[mt][j];
            v += __shfl_xor(v, 1, 32);
            v += __shfl_xor(v, 2, 32);
            v += __shfl_xor(v, 4, 32);
            v += __shfl_xor(v, 8, 32);
            if (r == 0) s_part[w][mt * 16 + grp * 8 + j] = v;
        }
    }
    __syncthreads();

    if (tid < M_BLK) {
        float s = 0.f;
        #pragma unroll
        for (int ww = 0; ww < WAVES; ++ww) s += s_part[ww][tid];
        s_scale[tid] = 1.0f / s;
    }
    __syncthreads();

    // --- normalized write-out (single pass over 128MB output) ---
    #pragma unroll
    for (int mt = 0; mt < 4; ++mt) {
        #pragma unroll
        for (int nt = 0; nt < 4; ++nt) {
            const int col = w * 64 + nt * 16 + r;
            #pragma unroll
            for (int j = 0; j < 8; ++j) {
                const int m = mt * 16 + grp * 8 + j;
                out[(size_t)(row0 + m) * K_CENT + col] = acc[mt][nt][j] * s_scale[m];
            }
        }
    }
}

// ---------------------------------------------------------------------------
extern "C" void kernel_launch(void* const* d_in, const int* in_sizes, int n_in,
                              void* d_out, int out_size, void* d_ws, size_t ws_size,
                              hipStream_t stream) {
    const float* features = (const float*)d_in[0];
    const float* centers  = (const float*)d_in[1];
    float* out = (float*)d_out;

    // workspace layout (~2.13 MB total)
    char* ws = (char*)d_ws;
    __bf16* ch = (__bf16*)ws;                                   // 1 MB
    __bf16* cl = (__bf16*)(ws + (size_t)(1 << 20));             // 1 MB
    float*  c2 = (float*)(ws + (size_t)(2 << 20));              // 4 KB
    float*  x2 = (float*)(ws + (size_t)(2 << 20) + 4096);       // 128 KB

    prep_centers<<<(K_CENT * 32) / 256, 256, 0, stream>>>(centers, ch, cl, c2);
    prep_x2<<<(N_ROWS * 32) / 256, 256, 0, stream>>>(features, x2);
    fuzzy_mu_kernel<<<N_ROWS / M_BLK, THREADS, 0, stream>>>(features, ch, cl, x2, c2, out);
}
